// MLP_39934605918750
// MI455X (gfx1250) — matrix operand order repaired and checked by reading, not compiled, not run on hardware
//
#include <hip/hip_runtime.h>
#include <hip/hip_bf16.h>

// ---------------------------------------------------------------------------
// MI455X (gfx1250, wave32) fused LeNet + MoE forward.
// All GEMM-shaped math (both convs via implicit-GEMM im2col, fc1, fc2,
// experts) runs on v_wmma_f32_16x16x32_f16. Intermediates live in the
// workspace (L2-resident: everything < 60MB vs 192MB L2). MoE top-2 combine
// is folded into the WMMA accumulation by scaling A-fragment rows (one token
// row per lane) with that lane's gate weight. Activation tiles are staged
// into LDS with the Tensor Data Mover (tensor_load_to_lds + s_wait_tensorcnt)
// so staging bypasses the VGPR/VMEM pipeline.
// ---------------------------------------------------------------------------

typedef __attribute__((ext_vector_type(16))) _Float16 v16h;
typedef __attribute__((ext_vector_type(8)))  _Float16 v8h;
typedef __attribute__((ext_vector_type(8)))  float    v8f;
typedef unsigned int u32x4 __attribute__((ext_vector_type(4)));
typedef int          i32x8 __attribute__((ext_vector_type(8)));
typedef int          i32x4 __attribute__((ext_vector_type(4)));

#define WMMA(a, b, c) \
  __builtin_amdgcn_wmma_f32_16x16x32_f16(false, (a), false, (b), (short)0, (c), false, false)

static __device__ __forceinline__ v16h make_a(v8h lo, v8h hi) {
  return __builtin_shufflevector(lo, hi, 0, 1, 2, 3, 4, 5, 6, 7, 8, 9, 10, 11, 12, 13, 14, 15);
}

// B fragment: two 8-half (16B) runs per lane from Wt[n][k] (ldb halves).
// lane<16: K=k0..k0+7 and k0+16..k0+23; lane>=16: K=k0+8..k0+15 and k0+24..k0+31; col n_base+lane%16.
static __device__ __forceinline__ v16h load_b_frag(const _Float16* __restrict__ Wt, int ldb,
                                                   int n_base, int k0, int lane) {
  const _Float16* p = Wt + (size_t)(n_base + (lane & 15)) * ldb + k0 + ((lane < 16) ? 0 : 8);
  return make_a(*(const v8h*)p, *(const v8h*)(p + 16));
}

// Tensor Data Mover: flat copy of `dwords` DWORDs from global -> LDS.
// D# per CDNA5 ISA ch.8: group0 = {count=1, lds_addr, global_addr, type=2},
// group1 = {data_size=4B, tensor_dim0=tile_dim0=dwords, dim1=1, stride0=dwords},
// groups 2/3 (and trailing group of the 6-arg builtin) zero (2D tensor).
// Tracked by TENSORcnt; EXEC ignored (one issue per wave executing it).
static __device__ __forceinline__ void tdm_load_to_lds(void* lds_dst, const void* gsrc,
                                                       unsigned dwords) {
  unsigned lds_off = (unsigned)(size_t)lds_dst;
  unsigned long long ga = (unsigned long long)gsrc;
  u32x4 g0;
  g0[0] = 1u;                                            // count=1 (valid descriptor)
  g0[1] = lds_off;                                       // bits 63:32  lds_addr
  g0[2] = (unsigned)(ga & 0xffffffffu);                  // bits 95:64  global_addr lo
  g0[3] = (unsigned)((ga >> 32) & 0x01ffffffu) | (2u << 30);  // addr hi | type=2
  i32x8 g1;
  g1[0] = (int)(2u << 16);                               // data_size=2 (4 bytes), mask=0
  g1[1] = (int)((dwords & 0xffffu) << 16);               // tensor_dim0[15:0] @ bits 63:48
  g1[2] = (int)((dwords >> 16) & 0xffffu) | (1 << 16);   // tensor_dim0 hi | tensor_dim1=1
  g1[3] = (int)((dwords & 0xffffu) << 16);               // tile_dim0 @ bits 127:112
  g1[4] = 1;                                             // tile_dim1=1 (tile_dim2=0)
  g1[5] = (int)dwords;                                   // tensor_dim0_stride lo
  g1[6] = 0;                                             // stride0 hi | stride1 lo
  g1[7] = 0;
  i32x4 gz4 = {0, 0, 0, 0};
  i32x8 gz8 = {0, 0, 0, 0, 0, 0, 0, 0};
  __builtin_amdgcn_tensor_load_to_lds(g0, g1, gz4, gz4, gz8, 0);
}

// ---------------------------------------------------------------------------
// Weight packing: fp32 -> f16, transposed-to-[n][k], zero padded.
// ---------------------------------------------------------------------------
__global__ __launch_bounds__(256) void pack_weights_kernel(
    const float* __restrict__ c1w, const float* __restrict__ c2w,
    const float* __restrict__ f1w, const float* __restrict__ f2w,
    const float* __restrict__ ew,
    _Float16* __restrict__ W1, _Float16* __restrict__ W2,
    _Float16* __restrict__ WF1, _Float16* __restrict__ WF2,
    _Float16* __restrict__ WE) {
  int idx = blockIdx.x * 256 + threadIdx.x;
  int stride = gridDim.x * 256;
  for (int i = idx; i < 16 * 96; i += stride) {       // conv1: [6][75] -> [16][96]
    int n = i / 96, k = i % 96;
    W1[i] = (n < 6 && k < 75) ? (_Float16)c1w[n * 75 + k] : (_Float16)0.f;
  }
  for (int i = idx; i < 16 * 160; i += stride) {      // conv2: [16][150] -> [16][160]
    int n = i / 160, k = i % 160;
    W2[i] = (n < 16 && k < 150) ? (_Float16)c2w[n * 150 + k] : (_Float16)0.f;
  }
  for (int i = idx; i < 128 * 416; i += stride) {     // fc1: [120][400] -> [128][416]
    int n = i / 416, k = i % 416;
    WF1[i] = (n < 120 && k < 400) ? (_Float16)f1w[n * 400 + k] : (_Float16)0.f;
  }
  for (int i = idx; i < 96 * 128; i += stride) {      // fc2: [84][120] -> [96][128]
    int n = i / 128, k = i % 128;
    WF2[i] = (n < 84 && k < 120) ? (_Float16)f2w[n * 120 + k] : (_Float16)0.f;
  }
  for (int i = idx; i < 8 * 96 * 96; i += stride) {   // experts: [8][84][84] -> [8][96][96]
    int e = i / 9216, r = i % 9216, n = r / 96, k = r % 96;
    WE[i] = (n < 84 && k < 84) ? (_Float16)ew[e * 7056 + n * 84 + k] : (_Float16)0.f;
  }
}

// ---------------------------------------------------------------------------
// conv1 (3->6, 5x5) + ReLU + 2x2 maxpool. One block per image.
// Implicit GEMM: M=784 pixels (49 tiles), K=75->96, N=16 (6 valid).
// (fp32 source needs conversion, so staged through VGPRs, not TDM.)
// ---------------------------------------------------------------------------
__global__ __launch_bounds__(256) void conv1_pool_kernel(
    const float* __restrict__ x, const float* __restrict__ conv1_b,
    const _Float16* __restrict__ Wt1, _Float16* __restrict__ act1) {
  __shared__ __align__(16) _Float16 img[4096];       // [3][32][32] + 1024 zero pad
  __shared__ _Float16 cout[6 * 784];                 // conv+relu result [ch][28*28]
  __shared__ int offt[96];                           // K -> image offset table
  const int tid = threadIdx.x, lane = tid & 31, wave = tid >> 5;
  const long n = blockIdx.x;

  const float* xin = x + n * 3072;
  for (int i = tid; i < 3072; i += 256) img[i] = (_Float16)xin[i];
  for (int i = tid; i < 1024; i += 256) img[3072 + i] = (_Float16)0.f;
  if (tid < 96) {
    int K = tid;
    offt[tid] = (K < 75) ? ((K / 25) * 1024 + ((K % 25) / 5) * 32 + (K % 5)) : 3072;
  }
  __syncthreads();

  const int m   = lane & 15;
  const int khA = (lane < 16) ? 0 : 8;
  const int ch  = lane & 15;
  const float bias = (ch < 6) ? conv1_b[ch] : 0.f;

  for (int tile = wave; tile < 49; tile += 8) {
    const int p = tile * 16 + m;                     // pixel id (always < 784)
    const int base = (p / 28) * 32 + (p % 28);
    v8f c = {};
#pragma unroll
    for (int ks = 0; ks < 3; ks++) {
      const int k0 = ks * 32;
      v16h a;
#pragma unroll
      for (int j = 0; j < 8; j++) {
        a[j]     = img[offt[k0 + khA + j] + base];
        a[j + 8] = img[offt[k0 + 16 + khA + j] + base];
      }
      v16h b = load_b_frag(Wt1, 96, 0, k0, lane);
      c = WMMA(a, b, c);
    }
    if (ch < 6) {
#pragma unroll
      for (int r = 0; r < 8; r++) {
        int pp = tile * 16 + r + ((lane < 16) ? 0 : 8);
        float v = c[r] + bias;
        cout[ch * 784 + pp] = (_Float16)(v > 0.f ? v : 0.f);
      }
    }
  }
  __syncthreads();

  _Float16* a1 = act1 + n * 1176;                    // [6][14][14]
  for (int i = tid; i < 1176; i += 256) {
    int cc = i / 196, rem = i % 196, y = rem / 14, xx = rem % 14;
    int b0 = cc * 784 + y * 56 + xx * 2;
    _Float16 v0 = cout[b0], v1 = cout[b0 + 1], v2 = cout[b0 + 28], v3 = cout[b0 + 29];
    _Float16 ma = v0 > v1 ? v0 : v1;
    _Float16 mb = v2 > v3 ? v2 : v3;
    a1[i] = ma > mb ? ma : mb;
  }
}

// ---------------------------------------------------------------------------
// conv2 (6->16, 5x5) + ReLU + 2x2 maxpool -> flattened [N][416] (400 + pad).
// Implicit GEMM: M=100 pixels (7 tiles, tail guarded), K=150->160, N=16.
// Input image staged LDS-direct via TDM (wave 0 issues one DMA).
// ---------------------------------------------------------------------------
__global__ __launch_bounds__(256) void conv2_pool_kernel(
    const _Float16* __restrict__ act1, const float* __restrict__ conv2_b,
    const _Float16* __restrict__ Wt2, _Float16* __restrict__ act2) {
  __shared__ __align__(16) _Float16 a1s[1176 + 168]; // [6][14][14] + zero pad
  __shared__ _Float16 c2o[16 * 100];                 // [ch][10*10]
  __shared__ int offt[160];
  const int tid = threadIdx.x, lane = tid & 31, wave = tid >> 5;
  const long n = blockIdx.x;

  if (wave == 0) {                                   // async DMA: 2352B image -> LDS
    tdm_load_to_lds(&a1s[0], act1 + n * 1176, 1176 * 2 / 4);
  }
  for (int i = tid; i < 168; i += 256) a1s[1176 + i] = (_Float16)0.f;
  if (tid < 160) {
    int K = tid;
    offt[tid] = (K < 150) ? ((K / 25) * 196 + ((K % 25) / 5) * 14 + (K % 5)) : 1176;
  }
  if (wave == 0) __builtin_amdgcn_s_wait_tensorcnt(0);
  __syncthreads();

  const int m   = lane & 15;
  const int khA = (lane < 16) ? 0 : 8;
  const int ch  = lane & 15;
  const float bias = conv2_b[ch];

  for (int tile = wave; tile < 7; tile += 8) {
    const int p = tile * 16 + m;                     // may exceed 99 in last tile
    const int base = (p / 10) * 14 + (p % 10);
    v8f c = {};
#pragma unroll
    for (int ks = 0; ks < 5; ks++) {
      const int k0 = ks * 32;
      v16h a;
#pragma unroll
      for (int j = 0; j < 8; j++) {
        a[j]     = a1s[offt[k0 + khA + j] + base];
        a[j + 8] = a1s[offt[k0 + 16 + khA + j] + base];
      }
      v16h b = load_b_frag(Wt2, 160, 0, k0, lane);
      c = WMMA(a, b, c);
    }
#pragma unroll
    for (int r = 0; r < 8; r++) {
      int pp = tile * 16 + r + ((lane < 16) ? 0 : 8);
      if (pp < 100) {
        float v = c[r] + bias;
        c2o[ch * 100 + pp] = (_Float16)(v > 0.f ? v : 0.f);
      }
    }
  }
  __syncthreads();

  _Float16* a2 = act2 + n * 416;                     // flatten [16][5][5] -> 400 (+16 pad)
  for (int i = tid; i < 416; i += 256) {
    if (i < 400) {
      int cc = i / 25, rem = i % 25, y = rem / 5, xx = rem % 5;
      int b0 = cc * 100 + y * 20 + xx * 2;
      _Float16 v0 = c2o[b0], v1 = c2o[b0 + 1], v2 = c2o[b0 + 10], v3 = c2o[b0 + 11];
      _Float16 ma = v0 > v1 ? v0 : v1;
      _Float16 mb = v2 > v3 ? v2 : v3;
      a2[i] = ma > mb ? ma : mb;
    } else {
      a2[i] = (_Float16)0.f;
    }
  }
}

// ---------------------------------------------------------------------------
// GEMM + bias + ReLU: C[M][ldc] = relu(A[M][K] @ Wt[Npad][K]^T + bias)
// One 16-row M-tile per wave; KS = K/32 k-steps, NT = Npad/16 accumulators.
// ---------------------------------------------------------------------------
template <int KS, int NT>
__global__ __launch_bounds__(256) void gemm_relu_kernel(
    const _Float16* __restrict__ A, int lda,
    const _Float16* __restrict__ Wt,
    const float* __restrict__ bias, int bias_n,
    _Float16* __restrict__ C, int ldc) {
  const int tid = threadIdx.x, lane = tid & 31, wave = tid >> 5;
  const long m_base = ((long)blockIdx.x * 8 + wave) * 16;
  const int khA = (lane < 16) ? 0 : 8;
  const int nn  = lane & 15;

  v8f acc[NT] = {};
  const _Float16* arow = A + (m_base + (lane & 15)) * lda + khA;
  __builtin_prefetch(arow, 0, 1);                    // global_prefetch_b8

  for (int ks = 0; ks < KS; ks++) {
    const _Float16* p = arow + ks * 32;
    v16h a = make_a(*(const v8h*)p, *(const v8h*)(p + 16));
#pragma unroll
    for (int nt = 0; nt < NT; nt++) {
      v16h b = load_b_frag(Wt, lda, nt * 16, ks * 32, lane);
      acc[nt] = WMMA(a, b, acc[nt]);
    }
  }
#pragma unroll
  for (int nt = 0; nt < NT; nt++) {
    const int col = nt * 16 + nn;
    const float bv = (col < bias_n) ? bias[col] : 0.f;
#pragma unroll
    for (int r = 0; r < 8; r++) {
      long row = m_base + r + ((lane < 16) ? 0 : 8);
      float v = acc[nt][r] + bv;
      C[row * ldc + col] = (_Float16)(v > 0.f ? v : 0.f);
    }
  }
}

// ---------------------------------------------------------------------------
// MoE (top-2 of 8 gating folded into WMMA accumulation) + head (84->10).
// Block = 128 tokens, one 16-token tile per wave; each wave TDM-loads its own
// 16x96 activation tile into its private LDS region.
// ---------------------------------------------------------------------------
__global__ __launch_bounds__(256) void moe_head_kernel(
    const _Float16* __restrict__ act4,   // [N][96] (84 valid + zero pad)
    const _Float16* __restrict__ We,     // [8][96][96]
    const float* __restrict__ gate_w,    // [84][8]
    const float* __restrict__ expert_b,  // [8][84]
    const float* __restrict__ head_w,    // [10][84]
    const float* __restrict__ head_b,    // [10]
    float* __restrict__ out) {           // [N][10]
  __shared__ __align__(16) _Float16 a4[8][16 * 96];  // per-wave token tile, reused for moe result
  __shared__ float gw[84 * 8];
  __shared__ float eb[8 * 84];
  __shared__ float hw[10 * 84];
  __shared__ float hb[10];
  __shared__ float comb[8][16][8];
  const int tid = threadIdx.x, lane = tid & 31, wave = tid >> 5;
  const long tok_base = (long)blockIdx.x * 128 + wave * 16;

  // async DMA of this wave's 16x96 f16 tile (3072B) into its LDS region;
  // overlaps with the fp32 weight staging below.
  tdm_load_to_lds(&a4[wave][0], act4 + tok_base * 96, 16 * 96 * 2 / 4);

  for (int i = tid; i < 672; i += 256) gw[i] = gate_w[i];
  for (int i = tid; i < 672; i += 256) eb[i] = expert_b[i];
  for (int i = tid; i < 840; i += 256) hw[i] = head_w[i];
  if (tid < 10) hb[tid] = head_b[tid];
  __syncthreads();
  __builtin_amdgcn_s_wait_tensorcnt(0);              // own tile resident

  _Float16* a4w = &a4[wave][0];
  const int t = lane & 15;

  // --- gate: logits, top-2, renormalized softmax over the top-2 ---
  float lg[8];
#pragma unroll
  for (int e = 0; e < 8; e++) lg[e] = 0.f;
  for (int k = 0; k < 84; k++) {
    float h = (float)a4w[t * 96 + k];
#pragma unroll
    for (int e = 0; e < 8; e++) lg[e] += h * gw[k * 8 + e];
  }
  int i1 = 0; float v1 = lg[0];
#pragma unroll
  for (int e = 1; e < 8; e++) if (lg[e] > v1) { v1 = lg[e]; i1 = e; }
  int i2 = 0; float v2 = -3.4e38f;
#pragma unroll
  for (int e = 0; e < 8; e++) if (e != i1 && lg[e] > v2) { v2 = lg[e]; i2 = e; }
  const float e2 = __expf(v2 - v1);
  const float p1 = 1.f / (1.f + e2), p2 = e2 / (1.f + e2);
  float cw[8];
#pragma unroll
  for (int e = 0; e < 8; e++) cw[e] = (e == i1) ? p1 : ((e == i2) ? p2 : 0.f);
  if (lane < 16) {
#pragma unroll
    for (int e = 0; e < 8; e++) comb[wave][t][e] = cw[e];
  }

  // --- A fragments for this tile (kept in regs; a4w is recycled later) ---
  const int khA = (lane < 16) ? 0 : 8;
  v16h afr[3];
#pragma unroll
  for (int ks = 0; ks < 3; ks++) {
    const _Float16* p = a4w + t * 96 + ks * 32 + khA;
    afr[ks] = make_a(*(const v8h*)p, *(const v8h*)(p + 16));
  }

  // --- experts: sum_e (cw[e] * h) @ W_e^T via scaled-A WMMA accumulation ---
  v8f acc[6] = {};
#pragma unroll
  for (int e = 0; e < 8; e++) {
    const _Float16 s = (_Float16)cw[e];
    v16h as0 = afr[0] * s, as1 = afr[1] * s, as2 = afr[2] * s;
    const _Float16* Be = We + e * 9216;
#pragma unroll
    for (int nt = 0; nt < 6; nt++) {
      acc[nt] = WMMA(as0, load_b_frag(Be, 96, nt * 16, 0, lane), acc[nt]);
      acc[nt] = WMMA(as1, load_b_frag(Be, 96, nt * 16, 32, lane), acc[nt]);
      acc[nt] = WMMA(as2, load_b_frag(Be, 96, nt * 16, 64, lane), acc[nt]);
    }
  }
  __syncthreads();                                    // comb visible to all lanes

  // --- combined expert bias, write moe result (f16) back into a4w ---
#pragma unroll
  for (int nt = 0; nt < 6; nt++) {
    const int col = nt * 16 + (lane & 15);
#pragma unroll
    for (int r = 0; r < 8; r++) {
      const int tr = r + ((lane < 16) ? 0 : 8);
      float v = 0.f;
      if (col < 84) {
        v = acc[nt][r];
#pragma unroll
        for (int e = 0; e < 8; e++) v += comb[wave][tr][e] * eb[e * 84 + col];
      }
      a4w[tr * 96 + col] = (_Float16)v;
    }
  }
  __syncthreads();                                    // moe result visible cross-lane

  // --- head: out[t][j] = sum_o moe[t][o]*head_w[j][o] + head_b[j] ---
  const int jbase = (lane < 16) ? 0 : 5;
  for (int j = jbase; j < jbase + 5; j++) {
    float s = hb[j];
    for (int o = 0; o < 84; o++) s += (float)a4w[t * 96 + o] * hw[j * 84 + o];
    out[(tok_base + t) * 10 + j] = s;
  }
}

// ---------------------------------------------------------------------------
extern "C" void kernel_launch(void* const* d_in, const int* in_sizes, int n_in,
                              void* d_out, int out_size, void* d_ws, size_t ws_size,
                              hipStream_t stream) {
  const float* x   = (const float*)d_in[0];
  const float* c1w = (const float*)d_in[1];
  const float* c1b = (const float*)d_in[2];
  const float* c2w = (const float*)d_in[3];
  const float* c2b = (const float*)d_in[4];
  const float* f1w = (const float*)d_in[5];
  const float* f1b = (const float*)d_in[6];
  const float* f2w = (const float*)d_in[7];
  const float* f2b = (const float*)d_in[8];
  const float* gw  = (const float*)d_in[9];
  const float* ew  = (const float*)d_in[10];
  const float* eb  = (const float*)d_in[11];
  const float* hw  = (const float*)d_in[12];
  const float* hb  = (const float*)d_in[13];
  float* out = (float*)d_out;

  const long N = (long)in_sizes[0] / 3072;  // 16384

  char* ws = (char*)d_ws;
  size_t off = 0;
  auto alloc = [&](size_t halves) -> _Float16* {
    _Float16* p = (_Float16*)(ws + off);
    off += ((halves * 2 + 255) / 256) * 256;
    return p;
  };
  _Float16* W1   = alloc(16 * 96);
  _Float16* W2   = alloc(16 * 160);
  _Float16* WF1  = alloc(128 * 416);
  _Float16* WF2  = alloc(96 * 128);
  _Float16* WE   = alloc(8 * 96 * 96);
  _Float16* act1 = alloc((size_t)N * 1176);
  _Float16* act2 = alloc((size_t)N * 416);
  _Float16* act3 = alloc((size_t)N * 128);
  _Float16* act4 = alloc((size_t)N * 96);
  (void)ws_size; (void)n_in; (void)out_size;

  pack_weights_kernel<<<64, 256, 0, stream>>>(c1w, c2w, f1w, f2w, ew, W1, W2, WF1, WF2, WE);
  conv1_pool_kernel<<<(int)N, 256, 0, stream>>>(x, c1b, W1, act1);
  conv2_pool_kernel<<<(int)N, 256, 0, stream>>>(act1, c2b, W2, act2);
  gemm_relu_kernel<13, 8><<<(int)(N / 128), 256, 0, stream>>>(act2, 416, WF1, f1b, 120, act3, 128);
  gemm_relu_kernel<4, 6><<<(int)(N / 128), 256, 0, stream>>>(act3, 128, WF2, f2b, 84, act4, 96);
  moe_head_kernel<<<(int)(N / 128), 256, 0, stream>>>(act4, WE, gw, eb, hw, hb, out);
}
